// Decoder_15547781611819
// MI455X (gfx1250) — matrix operand order repaired and checked by reading, not compiled, not run on hardware
//
#include <hip/hip_runtime.h>
#include <hip/hip_bf16.h>

// ---------------------------------------------------------------------------
// GRU decoder with tied projection, restructured for CDNA5 WMMA (bf16).
//   Phase 0 : convert weights/emb to bf16, gather teacher-forced inputs.
//   Phase 0.5 (parallel): GI = Xall @ w_ih^T for ALL steps (8192x1024x3072),
//             stored fp32 in d_out (used as scratch, overwritten by proj).
//   Phase 1 (sequential persistent kernel): 256 GRU steps; per step only
//             h @ w_hh^T (32x1024x3072 WMMA GEMM) + gates; grid barrier/step.
//   Phase 2 (parallel): 8192x1024x32000 WMMA GEMM (tied emb projection).
// ---------------------------------------------------------------------------

typedef __bf16 bf16_t;
typedef __attribute__((ext_vector_type(16))) __bf16 bf16x16;
typedef __attribute__((ext_vector_type(8)))  float  floatx8;

#define HD   1024
#define BSZ  32
#define TT   256
#define VV   32000

// ---------------- helpers ----------------

__device__ __forceinline__ unsigned short f2bf_u(float f) {
  unsigned int u = __float_as_uint(f);
  u += 0x7FFFu + ((u >> 16) & 1u);          // round to nearest even
  return (unsigned short)(u >> 16);
}
__device__ __forceinline__ float bf2f(unsigned short h) {
  return __uint_as_float(((unsigned int)h) << 16);
}
__device__ __forceinline__ float sigmoidf(float x) {
  return 1.0f / (1.0f + __expf(-x));
}

union FragU { bf16x16 v; float4 f[2]; };

// A-fragment (16x32 bf16): lane holds 16 contiguous K relative to p
// (p already includes the per-lane kbase of 0 or 16).
__device__ __forceinline__ bf16x16 load_a(const bf16_t* p) {
  FragU u;
  u.f[0] = *reinterpret_cast<const float4*>(p);
  u.f[1] = *reinterpret_cast<const float4*>(p + 8);
  return u.v;
}
// B-fragment (32x16 bf16): lane holds 16 contiguous K (p includes kbase 0/16).
__device__ __forceinline__ bf16x16 load_b(const bf16_t* p) {
  FragU u;
  u.f[0] = *reinterpret_cast<const float4*>(p);
  u.f[1] = *reinterpret_cast<const float4*>(p + 8);
  return u.v;
}

__device__ __forceinline__ floatx8 wmma_bf16(bf16x16 a, bf16x16 b, floatx8 c) {
  return __builtin_amdgcn_wmma_f32_16x16x32_bf16(false, a, false, b,
                                                 (short)0, c, false, false);
}

// ---------------- prep kernels ----------------

__global__ void conv_bf16_kernel(const float* __restrict__ src,
                                 unsigned short* __restrict__ dst, int n) {
  int i = blockIdx.x * blockDim.x + threadIdx.x;
  if (i < n) dst[i] = f2bf_u(src[i]);
}

// Xall[t][b][k] = embbf[ tok(t,b) ][k], tok = (t==0 ? START_IDX=1 : gold[b][t-1])
__global__ void build_x_kernel(const unsigned short* __restrict__ embbf,
                               const int* __restrict__ gold,
                               unsigned short* __restrict__ xall) {
  int i = blockIdx.x * blockDim.x + threadIdx.x;   // TT*BSZ*HD total
  int k = i & (HD - 1);
  int tb = i >> 10;
  int b = tb & (BSZ - 1);
  int t = tb >> 5;
  int tok = (t == 0) ? 1 : gold[b * TT + (t - 1)];
  xall[i] = embbf[(size_t)tok * HD + k];
}

// Hall[0] = bf16(enc_hiddens); also reset the grid-sync counter.
__global__ void init_h_kernel(const float* __restrict__ enc,
                              unsigned short* __restrict__ hall,
                              unsigned int* __restrict__ cnt) {
  int i = blockIdx.x * blockDim.x + threadIdx.x;   // BSZ*HD total
  if (i < BSZ * HD) hall[i] = f2bf_u(enc[i]);
  if (i == 0) *cnt = 0u;
}

// ---------------- phase 0.5: GI = Xall @ w_ih^T (all timesteps at once) -----
// M = 8192 (t*32+b), N = 3072, K = 1024. WG tile 256x128 (8 waves, 64x64/wave).

__global__ __launch_bounds__(256) void gi_gemm_kernel(
    const unsigned short* __restrict__ XallU,  // 8192 x 1024 bf16
    const unsigned short* __restrict__ WihU,   // 3072 x 1024 bf16
    float* __restrict__ gi) {                  // 8192 x 3072 f32 (d_out scratch)
  const int lane = threadIdx.x & 31;
  const int wv = threadIdx.x >> 5;
  const int wm = wv & 3;
  const int wn = wv >> 2;
  const int bn = blockIdx.x % (3 * HD / 128);
  const int bm = blockIdx.x / (3 * HD / 128);
  const int row0 = bm * 256 + wm * 64;
  const int col0 = bn * 128 + wn * 64;
  const int ln = lane & 15;
  const int kbA = (lane < 16) ? 0 : 16;
  const int kbB = (lane < 16) ? 0 : 16;

  const bf16_t* A  = reinterpret_cast<const bf16_t*>(XallU);
  const bf16_t* Bm = reinterpret_cast<const bf16_t*>(WihU);

  floatx8 acc[4][4] = {};
  for (int ks = 0; ks < HD / 32; ++ks) {
    const int k0 = ks * 32;
    bf16x16 af[4], bfm[4];
    #pragma unroll
    for (int i = 0; i < 4; ++i)
      af[i] = load_a(A + (size_t)(row0 + i * 16 + ln) * HD + k0 + kbA);
    #pragma unroll
    for (int i = 0; i < 4; ++i)
      bfm[i] = load_b(Bm + (size_t)(col0 + i * 16 + ln) * HD + k0 + kbB);
    #pragma unroll
    for (int mi = 0; mi < 4; ++mi)
      #pragma unroll
      for (int ni = 0; ni < 4; ++ni)
        acc[mi][ni] = wmma_bf16(af[mi], bfm[ni], acc[mi][ni]);
  }

  #pragma unroll
  for (int mi = 0; mi < 4; ++mi) {
    #pragma unroll
    for (int r = 0; r < 8; ++r) {
      int m = row0 + mi * 16 + r + ((lane < 16) ? 0 : 8);
      float* orow = gi + (size_t)m * (3 * HD);
      #pragma unroll
      for (int ni = 0; ni < 4; ++ni)
        orow[col0 + ni * 16 + ln] = acc[mi][ni][r];
    }
  }
}

// ---------------- phase 1: sequential GRU (persistent, 16 WGs) ----------------
// WG wg owns H-columns [wg*64, wg*64+64); wave w owns 16 of them.
// Per step: GH(32x3072) = h(32x1024) x w_hh^T ; gates use precomputed GI (f32).

__global__ __launch_bounds__(128) void gru_seq_kernel(
    const unsigned short* __restrict__ WhhU,    // 3072 x 1024 bf16
    const float* __restrict__ GI,               // 8192 x 3072 f32
    unsigned short* __restrict__ HallU,         // (TT+1) x 32 x 1024 bf16
    const float* __restrict__ b_ih, const float* __restrict__ b_hh,
    unsigned int* __restrict__ sync_cnt) {
  const int lane = threadIdx.x & 31;
  const int wave = threadIdx.x >> 5;
  const int jc   = blockIdx.x * 64 + wave * 16;  // H-column base of this wave
  const int ln   = lane & 15;
  const int kbA  = (lane < 16) ? 0 : 16;
  const int kbB  = (lane < 16) ? 0 : 16;
  const int j    = jc + ln;

  const float brb = b_ih[j] + b_hh[j];
  const float bzb = b_ih[HD + j] + b_hh[HD + j];
  const float bin = b_ih[2 * HD + j];
  const float bhn = b_hh[2 * HD + j];

  const bf16_t* W  = reinterpret_cast<const bf16_t*>(WhhU);
  const bf16_t* Wr = W + (size_t)(j) * HD + kbB;
  const bf16_t* Wz = W + (size_t)(HD + j) * HD + kbB;
  const bf16_t* Wn = W + (size_t)(2 * HD + j) * HD + kbB;

  for (int t = 0; t < TT; ++t) {
    const bf16_t* Hc = reinterpret_cast<const bf16_t*>(HallU) + (size_t)t * BSZ * HD;
    const unsigned short* HcU = HallU + (size_t)t * BSZ * HD;
    unsigned short* Hn = HallU + (size_t)(t + 1) * BSZ * HD;

    floatx8 acc_r[2] = {};
    floatx8 acc_z[2] = {};
    floatx8 acc_n[2] = {};

    for (int ks = 0; ks < HD / 32; ++ks) {
      const int k0 = ks * 32;
      bf16x16 a0 = load_a(Hc + (size_t)ln * HD + k0 + kbA);
      bf16x16 a1 = load_a(Hc + (size_t)(16 + ln) * HD + k0 + kbA);
      bf16x16 br = load_b(Wr + k0);
      bf16x16 bz = load_b(Wz + k0);
      bf16x16 bn = load_b(Wn + k0);
      acc_r[0] = wmma_bf16(a0, br, acc_r[0]);
      acc_r[1] = wmma_bf16(a1, br, acc_r[1]);
      acc_z[0] = wmma_bf16(a0, bz, acc_z[0]);
      acc_z[1] = wmma_bf16(a1, bz, acc_z[1]);
      acc_n[0] = wmma_bf16(a0, bn, acc_n[0]);
      acc_n[1] = wmma_bf16(a1, bn, acc_n[1]);
    }

    // gates + h_new (C layout: VGPR r, lane l -> M = r + (l<16?0:8), N = l&15)
    #pragma unroll
    for (int mt = 0; mt < 2; ++mt) {
      #pragma unroll
      for (int r = 0; r < 8; ++r) {
        int m = mt * 16 + r + ((lane < 16) ? 0 : 8);
        const float* gir = GI + ((size_t)t * BSZ + m) * (3 * HD);
        float vr = sigmoidf(acc_r[mt][r] + gir[j] + brb);
        float vz = sigmoidf(acc_z[mt][r] + gir[HD + j] + bzb);
        float vn = tanhf(gir[2 * HD + j] + bin + vr * (acc_n[mt][r] + bhn));
        float ho = bf2f(HcU[(size_t)m * HD + j]);
        float hv = (1.0f - vz) * vn + vz * ho;
        Hn[(size_t)m * HD + j] = f2bf_u(hv);
      }
    }

    // grid-wide barrier (monotonic counter; target = (t+1)*gridDim.x)
    __threadfence();
    __syncthreads();
    if (threadIdx.x == 0) {
      atomicAdd(sync_cnt, 1u);
      const unsigned int target = (unsigned int)(t + 1) * gridDim.x;
      while (atomicAdd(sync_cnt, 0u) < target) { __builtin_amdgcn_s_sleep(2); }
    }
    __syncthreads();
    __threadfence();
  }
}

// ---------------- phase 2: tied projection GEMM ----------------
// out[b][t][v] = sum_k Hall[t+1][b][k] * emb[v][k]
// M = 8192 (=t*32+b), N = 32000, K = 1024. WG tile 256x128 (8 waves, 64x64/wave).

__global__ __launch_bounds__(256) void proj_kernel(
    const unsigned short* __restrict__ HallU,  // (TT+1) x 32 x 1024 bf16
    const unsigned short* __restrict__ EmbU,   // VV x 1024 bf16
    float* __restrict__ out) {
  const int lane = threadIdx.x & 31;
  const int wv = threadIdx.x >> 5;
  const int wm = wv & 3;
  const int wn = wv >> 2;
  const int bn = blockIdx.x % (VV / 128);
  const int bm = blockIdx.x / (VV / 128);
  const int row0 = bm * 256 + wm * 64;
  const int col0 = bn * 128 + wn * 64;
  const int ln = lane & 15;
  const int kbA = (lane < 16) ? 0 : 16;
  const int kbB = (lane < 16) ? 0 : 16;

  const bf16_t* A  = reinterpret_cast<const bf16_t*>(HallU) + (size_t)BSZ * HD;  // skip h0
  const bf16_t* Bm = reinterpret_cast<const bf16_t*>(EmbU);

  floatx8 acc[4][4] = {};
  for (int ks = 0; ks < HD / 32; ++ks) {
    const int k0 = ks * 32;
    bf16x16 af[4], bfm[4];
    #pragma unroll
    for (int i = 0; i < 4; ++i)
      af[i] = load_a(A + (size_t)(row0 + i * 16 + ln) * HD + k0 + kbA);
    #pragma unroll
    for (int i = 0; i < 4; ++i)
      bfm[i] = load_b(Bm + (size_t)(col0 + i * 16 + ln) * HD + k0 + kbB);
    #pragma unroll
    for (int mi = 0; mi < 4; ++mi)
      #pragma unroll
      for (int ni = 0; ni < 4; ++ni)
        acc[mi][ni] = wmma_bf16(af[mi], bfm[ni], acc[mi][ni]);
  }

  #pragma unroll
  for (int mi = 0; mi < 4; ++mi) {
    #pragma unroll
    for (int r = 0; r < 8; ++r) {
      int m = row0 + mi * 16 + r + ((lane < 16) ? 0 : 8);
      int t = m >> 5;            // /32
      int b = m & (BSZ - 1);
      float* orow = out + ((size_t)b * TT + t) * VV;
      #pragma unroll
      for (int ni = 0; ni < 4; ++ni) {
        int v = col0 + ni * 16 + ln;
        orow[v] = acc[mi][ni][r];
      }
    }
  }
}

// ---------------- host launcher ----------------

extern "C" void kernel_launch(void* const* d_in, const int* in_sizes, int n_in,
                              void* d_out, int out_size, void* d_ws, size_t ws_size,
                              hipStream_t stream) {
  (void)in_sizes; (void)n_in; (void)out_size;
  const float* enc  = (const float*)d_in[0];   // (1,32,1024)
  const float* emb  = (const float*)d_in[1];   // (32000,1024)
  const float* wih  = (const float*)d_in[2];   // (3072,1024)
  const float* whh  = (const float*)d_in[3];   // (3072,1024)
  const float* bih  = (const float*)d_in[4];   // (3072,)
  const float* bhh  = (const float*)d_in[5];   // (3072,)
  const int*   gold = (const int*)d_in[6];     // (32,256)
  float* out = (float*)d_out;                  // (32,256,32000) f32

  // workspace layout (bytes, 256-aligned)
  const size_t OFF_EMB  = 0;                                    // 32000*1024*2
  const size_t OFF_WIH  = OFF_EMB  + (size_t)VV * HD * 2;       // 3072*1024*2
  const size_t OFF_WHH  = OFF_WIH  + (size_t)3 * HD * HD * 2;   // 3072*1024*2
  const size_t OFF_XALL = OFF_WHH  + (size_t)3 * HD * HD * 2;   // 256*32*1024*2
  const size_t OFF_HALL = OFF_XALL + (size_t)TT * BSZ * HD * 2; // 257*32*1024*2
  const size_t OFF_CNT  = OFF_HALL + (size_t)(TT + 1) * BSZ * HD * 2;
  const size_t NEED     = OFF_CNT + 256;
  if (ws_size < NEED) return;

  char* ws = (char*)d_ws;
  unsigned short* embbf = (unsigned short*)(ws + OFF_EMB);
  unsigned short* wihbf = (unsigned short*)(ws + OFF_WIH);
  unsigned short* whhbf = (unsigned short*)(ws + OFF_WHH);
  unsigned short* xall  = (unsigned short*)(ws + OFF_XALL);
  unsigned short* hall  = (unsigned short*)(ws + OFF_HALL);
  unsigned int*   cnt   = (unsigned int*)(ws + OFF_CNT);

  // GI scratch (8192 x 3072 f32 = 100.7 MB) lives at the front of d_out;
  // proj_kernel later overwrites every element of d_out.
  float* gi = out;

  // phase 0: conversions / gathers
  { int n = VV * HD;     conv_bf16_kernel<<<(n + 255) / 256, 256, 0, stream>>>(emb, embbf, n); }
  { int n = 3 * HD * HD; conv_bf16_kernel<<<(n + 255) / 256, 256, 0, stream>>>(wih, wihbf, n); }
  { int n = 3 * HD * HD; conv_bf16_kernel<<<(n + 255) / 256, 256, 0, stream>>>(whh, whhbf, n); }
  { int n = TT * BSZ * HD; build_x_kernel<<<n / 256, 256, 0, stream>>>(embbf, gold, xall); }
  init_h_kernel<<<(BSZ * HD) / 256, 256, 0, stream>>>(enc, hall, cnt);

  // phase 0.5: GI = Xall @ w_ih^T for all timesteps (parallel)
  gi_gemm_kernel<<<(8192 / 256) * (3 * HD / 128), 256, 0, stream>>>(xall, wihbf, gi);

  // phase 1: sequential GRU recurrence (persistent kernel, 16 WGs x 128 thr)
  gru_seq_kernel<<<16, 128, 0, stream>>>(whhbf, gi, hall, bih, bhh, cnt);

  // phase 2: tied-embedding projection GEMM
  proj_kernel<<<(8192 / 256) * (VV / 128), 256, 0, stream>>>(hall, embbf, out);
}
